// MultiHeadedAttention_32280974196997
// MI455X (gfx1250) — matrix operand order repaired and checked by reading, not compiled, not run on hardware
//
#include <hip/hip_runtime.h>
#include <hip/hip_bf16.h>

typedef __attribute__((ext_vector_type(16))) _Float16 v16h;
typedef __attribute__((ext_vector_type(8)))  _Float16 v8h;
typedef __attribute__((ext_vector_type(4)))  _Float16 v4h;
typedef __attribute__((ext_vector_type(8)))  float    v8f;

#define BB 2
#define TT 1024
#define DD 1024
#define HH 16
#define DKK 64
#define INV_SQRT_DK 0.125f

#define WMMA_F32_F16(a, b, c) \
  __builtin_amdgcn_wmma_f32_16x16x32_f16(false, (a), false, (b), (short)0, (c), false, false)

// ---- Fragment loaders (CDNA5 16x16x32 f16 VGPR layouts, CDNA5 ISA) ----
// A (16x32, MxK), tile row-major [m][k], K contiguous.
__device__ __forceinline__ v16h frag_a(const _Float16* tile, int stride) {
  const int lane = threadIdx.x & 31;
  const int row  = lane & 15;
  const int kb   = (lane < 16) ? 0 : 8;
  const _Float16* p = tile + row * stride + kb;
  v8h lo = *(const v8h*)(p);
  v8h hi = *(const v8h*)(p + 16);
  v16h a;
#pragma unroll
  for (int i = 0; i < 8; ++i) { a[i] = lo[i]; a[8 + i] = hi[i]; }
  return a;
}

// B (32x16, KxN), staged TRANSPOSED in LDS as [n][k] so K is contiguous.
__device__ __forceinline__ v16h frag_b(const _Float16* tileT, int stride) {
  const int lane = threadIdx.x & 31;
  const int n    = lane & 15;
  const int kb   = (lane < 16) ? 0 : 8;
  const _Float16* p = tileT + n * stride + kb;
  v8h lo = *(const v8h*)(p);
  v8h hi = *(const v8h*)(p + 16);
  v16h b;
#pragma unroll
  for (int i = 0; i < 8; ++i) { b[i] = lo[i]; b[8 + i] = hi[i]; }
  return b;
}

__device__ __forceinline__ v4h cvt4(float4 f) {
  v4h h;
  h[0] = (_Float16)f.x; h[1] = (_Float16)f.y;
  h[2] = (_Float16)f.z; h[3] = (_Float16)f.w;
  return h;
}

// ---------------------------------------------------------------------------
// Kernel 1/7: projection GEMM, M=2048, N=K=1024, 64x64 tile, 128 thr (4 waves).
// float4 global staging + register prefetch pipeline overlapping WMMA.
// mode 0: outH[(b,h,t,dk)] = ((X@W)+bias)*scale as f16
// mode 1: outF[m*N+n]      = (X@W)+bias as f32
// ---------------------------------------------------------------------------
__global__ __launch_bounds__(128) void gemm_proj(
    const float* __restrict__ X, const float* __restrict__ W,
    const float* __restrict__ bias, float scale, int mode,
    _Float16* __restrict__ outH, float* __restrict__ outF) {
  __shared__ __align__(16) _Float16 As[64 * 32];   // [m][k]
  __shared__ __align__(16) _Float16 BsT[64 * 32];  // [n][k]
  const int tid = threadIdx.x, wave = tid >> 5, lane = tid & 31;
  const int m0 = blockIdx.x * 64, n0 = blockIdx.y * 64;
  const int K = DD, N = DD;
  float4 ra[4], rb[4];
  auto ld = [&](int kk) {
#pragma unroll
    for (int i = 0; i < 4; ++i) {
      int e = (i * 128 + tid) * 4;
      int m = e >> 5, k = e & 31;
      ra[i] = *(const float4*)&X[(size_t)(m0 + m) * K + kk + k];
    }
#pragma unroll
    for (int i = 0; i < 4; ++i) {
      int e = (i * 128 + tid) * 4;
      int k = e >> 6, n = e & 63;
      rb[i] = *(const float4*)&W[(size_t)(kk + k) * N + n0 + n];
    }
  };
  auto st = [&]() {
#pragma unroll
    for (int i = 0; i < 4; ++i) {
      int e = (i * 128 + tid) * 4;
      *(v4h*)&As[e] = cvt4(ra[i]);
    }
#pragma unroll
    for (int i = 0; i < 4; ++i) {
      int e = (i * 128 + tid) * 4;
      int k = e >> 6, n = e & 63;
      v4h h = cvt4(rb[i]);
#pragma unroll
      for (int j = 0; j < 4; ++j) BsT[(n + j) * 32 + k] = h[j];
    }
  };
  v8f acc[4] = {};
  ld(0);
  for (int kk = 0; kk < K; kk += 32) {
    st();
    __syncthreads();
    if (kk + 32 < K) ld(kk + 32);
    v16h a = frag_a(&As[wave * 16 * 32], 32);
#pragma unroll
    for (int nt = 0; nt < 4; ++nt) {
      v16h bf = frag_b(&BsT[nt * 16 * 32], 32);
      acc[nt] = WMMA_F32_F16(a, bf, acc[nt]);
    }
    __syncthreads();
  }
  const int roff = (lane < 16) ? 0 : 8, ncol = lane & 15;
#pragma unroll
  for (int nt = 0; nt < 4; ++nt)
#pragma unroll
    for (int r = 0; r < 8; ++r) {
      int m = m0 + wave * 16 + roff + r;
      int n = n0 + nt * 16 + ncol;
      float v = (acc[nt][r] + bias[n]) * scale;
      if (mode == 0) {
        int b = m >> 10, t = m & (TT - 1), h = n >> 6, dk = n & 63;
        outH[(((size_t)(b * HH + h) * TT + t) << 6) + dk] = (_Float16)v;
      } else {
        outF[(size_t)m * N + n] = v;
      }
    }
}

// ---------------------------------------------------------------------------
// Kernel 4: S[bh,t,s] = q.k^T + rel_bias[h,t,s].  64x64 tile, K=64.
// ---------------------------------------------------------------------------
__global__ __launch_bounds__(128) void scores_qk(
    const _Float16* __restrict__ qh, const _Float16* __restrict__ kh,
    const float* __restrict__ relb, float* __restrict__ S) {
  __shared__ __align__(16) _Float16 Qs[64 * 64];
  __shared__ __align__(16) _Float16 Ks[64 * 64];
  const int tid = threadIdx.x, wave = tid >> 5, lane = tid & 31;
  const int blk = blockIdx.x;
  const int bh = blk >> 8;
  const int tt = ((blk >> 4) & 15) * 64;
  const int ss = (blk & 15) * 64;
#pragma unroll
  for (int i = 0; i < 4; ++i) {
    int e = (i * 128 + tid) * 8;
    int r = e >> 6, d = e & 63;
    *(v8h*)&Qs[e] = *(const v8h*)&qh[((size_t)(bh * TT + tt + r) << 6) + d];
    *(v8h*)&Ks[e] = *(const v8h*)&kh[((size_t)(bh * TT + ss + r) << 6) + d];
  }
  __syncthreads();
  v8f acc[4] = {};
#pragma unroll
  for (int kk = 0; kk < 64; kk += 32) {
    v16h a = frag_a(&Qs[(wave * 16) * 64 + kk], 64);
#pragma unroll
    for (int nt = 0; nt < 4; ++nt) {
      v16h bf = frag_b(&Ks[(nt * 16) * 64 + kk], 64);
      acc[nt] = WMMA_F32_F16(a, bf, acc[nt]);
    }
  }
  const int roff = (lane < 16) ? 0 : 8, ncol = lane & 15;
  const int h = bh & (HH - 1);
#pragma unroll
  for (int nt = 0; nt < 4; ++nt)
#pragma unroll
    for (int r = 0; r < 8; ++r) {
      int t = tt + wave * 16 + roff + r;
      int s = ss + nt * 16 + ncol;
      S[((size_t)bh * TT + t) * TT + s] =
          acc[nt][r] + relb[((size_t)h * TT + t) * TT + s];
    }
}

// ---------------------------------------------------------------------------
// Kernel 5: one block per t.  S[bh,t,s] += q[bh,t,:].pos_k[t,s,:]
// M=32 (bh), N=1024 (s in 64-blocks), K=64; pos_k[t] streamed once, prefetched.
// ---------------------------------------------------------------------------
__global__ __launch_bounds__(128) void scores_bp(
    const _Float16* __restrict__ qh, const float* __restrict__ pos_k,
    float* __restrict__ S) {
  __shared__ __align__(16) _Float16 Qs[32 * 64];  // [bh][d]
  __shared__ __align__(16) _Float16 Ps[64 * 64];  // [s][d] == B-transposed
  const int tid = threadIdx.x, wave = tid >> 5, lane = tid & 31;
  const int t = blockIdx.x;
#pragma unroll
  for (int i = 0; i < 2; ++i) {
    int e = (i * 128 + tid) * 8;
    int r = e >> 6, d = e & 63;
    *(v8h*)&Qs[e] = *(const v8h*)&qh[((size_t)(r * TT + t) << 6) + d];
  }
  float4 rp[8];
  auto ldP = [&](int sb) {
#pragma unroll
    for (int i = 0; i < 8; ++i) {
      int e = (i * 128 + tid) * 4;  // tile is fully contiguous in memory
      rp[i] = *(const float4*)&pos_k[(((size_t)t * TT + sb) << 6) + e];
    }
  };
  const int mrow = (wave & 1) * 16;
  const int nbase = (wave >> 1) * 32;
  const int roff = (lane < 16) ? 0 : 8, ncol = lane & 15;
  ldP(0);
  for (int sb = 0; sb < TT; sb += 64) {
    __syncthreads();
#pragma unroll
    for (int i = 0; i < 8; ++i) {
      int e = (i * 128 + tid) * 4;
      *(v4h*)&Ps[e] = cvt4(rp[i]);
    }
    __syncthreads();
    if (sb + 64 < TT) ldP(sb + 64);
    v8f acc[2] = {};
#pragma unroll
    for (int kk = 0; kk < 64; kk += 32) {
      v16h a = frag_a(&Qs[mrow * 64 + kk], 64);
#pragma unroll
      for (int j = 0; j < 2; ++j) {
        v16h bf = frag_b(&Ps[(nbase + j * 16) * 64 + kk], 64);
        acc[j] = WMMA_F32_F16(a, bf, acc[j]);
      }
    }
#pragma unroll
    for (int j = 0; j < 2; ++j)
#pragma unroll
      for (int r = 0; r < 8; ++r) {
        int bh = mrow + roff + r;
        int s = sb + nbase + j * 16 + ncol;
        S[((size_t)bh * TT + t) * TT + s] += acc[j][r];
      }
  }
}

// ---------------------------------------------------------------------------
// Kernel 6: masked softmax per (bh,t) row, single pass over S.
// 256 threads x 4 contiguous elements; row lives in registers.
// ---------------------------------------------------------------------------
__global__ __launch_bounds__(256) void softmax_rows(
    const float* __restrict__ S, const int* __restrict__ mask,
    _Float16* __restrict__ attn) {
  __shared__ float red[256];
  const int tid = threadIdx.x;
  const int row = blockIdx.x;             // bh*T + t
  const int bh = row >> 10, t = row & (TT - 1);
  const int b = bh / HH;
  const float4 sv = *(const float4*)(S + (size_t)row * TT + tid * 4);
  const int4  mv = *(const int4*)(mask + ((size_t)b * TT + t) * TT + tid * 4);
  const float NEG = -3.4e38f;
  float s0 = (mv.x != 0) ? sv.x : NEG;
  float s1 = (mv.y != 0) ? sv.y : NEG;
  float s2 = (mv.z != 0) ? sv.z : NEG;
  float s3 = (mv.w != 0) ? sv.w : NEG;
  float mx = fmaxf(fmaxf(s0, s1), fmaxf(s2, s3));
  red[tid] = mx; __syncthreads();
  for (int w = 128; w > 0; w >>= 1) {
    if (tid < w) red[tid] = fmaxf(red[tid], red[tid + w]);
    __syncthreads();
  }
  mx = red[0]; __syncthreads();
  float e0 = (mv.x != 0) ? __expf(s0 - mx) : 0.f;
  float e1 = (mv.y != 0) ? __expf(s1 - mx) : 0.f;
  float e2 = (mv.z != 0) ? __expf(s2 - mx) : 0.f;
  float e3 = (mv.w != 0) ? __expf(s3 - mx) : 0.f;
  red[tid] = e0 + e1 + e2 + e3; __syncthreads();
  for (int w = 128; w > 0; w >>= 1) {
    if (tid < w) red[tid] += red[tid + w];
    __syncthreads();
  }
  const float sum = red[0];
  const float inv = (sum > 0.f) ? (1.f / sum) : 0.f;
  v4h h;
  h[0] = (_Float16)(e0 * inv); h[1] = (_Float16)(e1 * inv);
  h[2] = (_Float16)(e2 * inv); h[3] = (_Float16)(e3 * inv);
  *(v4h*)(attn + (size_t)row * TT + tid * 4) = h;
}

// ---------------------------------------------------------------------------
// Kernel 7: x[bh,t,:] = attn[bh,t,:] @ v[bh]  (64 t-rows, N=64, K=1024),
// register-prefetch pipeline; writes xacc (B,T,D) with '='.
// ---------------------------------------------------------------------------
__global__ __launch_bounds__(128) void attn_v_kernel(
    const _Float16* __restrict__ attn, const _Float16* __restrict__ vh,
    float* __restrict__ xacc) {
  __shared__ __align__(16) _Float16 As[64 * 32];   // [t][s]
  __shared__ __align__(16) _Float16 VsT[64 * 32];  // [d][s]
  const int tid = threadIdx.x, wave = tid >> 5, lane = tid & 31;
  const int bh = blockIdx.x >> 4;
  const int tt = (blockIdx.x & 15) * 64;
  v8h rA[2], rV[2];
  auto ld = [&](int kk) {
#pragma unroll
    for (int i = 0; i < 2; ++i) {
      int e = (i * 128 + tid) * 8;
      int r = e >> 5, s = e & 31;
      rA[i] = *(const v8h*)&attn[((size_t)bh * TT + tt + r) * TT + kk + s];
      int s2 = e >> 6, d = e & 63;
      rV[i] = *(const v8h*)&vh[((size_t)(bh * TT + kk + s2) << 6) + d];
    }
  };
  auto st = [&]() {
#pragma unroll
    for (int i = 0; i < 2; ++i) {
      int e = (i * 128 + tid) * 8;
      *(v8h*)&As[e] = rA[i];
      int s2 = e >> 6, d = e & 63;
#pragma unroll
      for (int j = 0; j < 8; ++j) VsT[(d + j) * 32 + s2] = rV[i][j];
    }
  };
  v8f acc[4] = {};
  ld(0);
  for (int kk = 0; kk < TT; kk += 32) {
    st();
    __syncthreads();
    if (kk + 32 < TT) ld(kk + 32);
    v16h a = frag_a(&As[wave * 16 * 32], 32);
#pragma unroll
    for (int nt = 0; nt < 4; ++nt) {
      v16h bf = frag_b(&VsT[nt * 16 * 32], 32);
      acc[nt] = WMMA_F32_F16(a, bf, acc[nt]);
    }
    __syncthreads();
  }
  const int roff = (lane < 16) ? 0 : 8, ncol = lane & 15;
  const int b = bh / HH, h = bh & (HH - 1);
#pragma unroll
  for (int nt = 0; nt < 4; ++nt)
#pragma unroll
    for (int r = 0; r < 8; ++r) {
      int t = tt + wave * 16 + roff + r;
      int d = nt * 16 + ncol;
      xacc[((size_t)(b * TT + t) << 10) + (h << 6) + d] = acc[nt][r];
    }
}

// ---------------------------------------------------------------------------
// Kernel 8: one block per t.  xacc[b,t,h*64+d] += attn[bh,t,:].pos_v[t,:,d]
// M=32 (bh), N=64 (d), K=1024 (s); pos_v[t] streamed once, prefetched.
// ---------------------------------------------------------------------------
__global__ __launch_bounds__(128) void attn_posv_kernel(
    const _Float16* __restrict__ attn, const float* __restrict__ pos_v,
    float* __restrict__ xacc) {
  __shared__ __align__(16) _Float16 As[32 * 32];   // [bh][s]
  __shared__ __align__(16) _Float16 PsT[64 * 32];  // [d][s]
  const int tid = threadIdx.x, wave = tid >> 5, lane = tid & 31;
  const int t = blockIdx.x;
  v8h rA;
  float4 rP[4];
  auto ld = [&](int kk) {
    int e = tid * 8;
    rA = *(const v8h*)&attn[((size_t)(e >> 5) * TT + t) * TT + kk + (e & 31)];
#pragma unroll
    for (int i = 0; i < 4; ++i) {
      int e2 = (i * 128 + tid) * 4;
      int s = e2 >> 6, d = e2 & 63;
      rP[i] = *(const float4*)&pos_v[(((size_t)t * TT + kk + s) << 6) + d];
    }
  };
  auto st = [&]() {
    int e = tid * 8;
    *(v8h*)&As[e] = rA;
#pragma unroll
    for (int i = 0; i < 4; ++i) {
      int e2 = (i * 128 + tid) * 4;
      int s = e2 >> 6, d = e2 & 63;
      v4h h = cvt4(rP[i]);
#pragma unroll
      for (int j = 0; j < 4; ++j) PsT[(d + j) * 32 + s] = h[j];
    }
  };
  const int mrow = (wave & 1) * 16;
  const int nbase = (wave >> 1) * 32;
  v8f acc[2] = {};
  ld(0);
  for (int kk = 0; kk < TT; kk += 32) {
    st();
    __syncthreads();
    if (kk + 32 < TT) ld(kk + 32);
    v16h a = frag_a(&As[mrow * 32], 32);
#pragma unroll
    for (int j = 0; j < 2; ++j) {
      v16h bf = frag_b(&PsT[(nbase + j * 16) * 32], 32);
      acc[j] = WMMA_F32_F16(a, bf, acc[j]);
    }
    __syncthreads();
  }
  const int roff = (lane < 16) ? 0 : 8, ncol = lane & 15;
#pragma unroll
  for (int j = 0; j < 2; ++j)
#pragma unroll
    for (int r = 0; r < 8; ++r) {
      int bh = mrow + roff + r;
      int b = bh / HH, h = bh & (HH - 1);
      int d = nbase + j * 16 + ncol;
      xacc[((size_t)(b * TT + t) << 10) + (h << 6) + d] += acc[j][r];
    }
}

// ---------------------------------------------------------------------------
extern "C" void kernel_launch(void* const* d_in, const int* in_sizes, int n_in,
                              void* d_out, int out_size, void* d_ws, size_t ws_size,
                              hipStream_t stream) {
  (void)in_sizes; (void)n_in; (void)out_size; (void)ws_size;
  const float* query = (const float*)d_in[0];
  const float* key_  = (const float*)d_in[1];
  const float* value = (const float*)d_in[2];
  const float* pos_k = (const float*)d_in[3];
  const float* pos_v = (const float*)d_in[4];
  const int*   mask  = (const int*)d_in[5];
  const float* relb  = (const float*)d_in[6];
  const float* Wq = (const float*)d_in[7];   const float* bq = (const float*)d_in[8];
  const float* Wk = (const float*)d_in[9];   const float* bk = (const float*)d_in[10];
  const float* Wv = (const float*)d_in[11];  const float* bv = (const float*)d_in[12];
  const float* Wo = (const float*)d_in[13];  const float* bo = (const float*)d_in[14];

  // Workspace layout (bytes):
  //   [0,4Mi)    qh   f16 (B,H,T,DK)
  //   [4,8Mi)    kh   f16
  //   [8,12Mi)   vh   f16
  //   [12,20Mi)  xacc f32 (B,T,D)
  //   [20,84Mi)  attn f16 (B*H,T,T)
  //   [84,212Mi) S    f32 (B*H,T,T)
  char* ws = (char*)d_ws;
  _Float16* qh   = (_Float16*)(ws);
  _Float16* kh   = (_Float16*)(ws + ((size_t)4  << 20));
  _Float16* vh   = (_Float16*)(ws + ((size_t)8  << 20));
  float*    xacc = (float*)   (ws + ((size_t)12 << 20));
  _Float16* attn = (_Float16*)(ws + ((size_t)20 << 20));
  float*    S    = (float*)   (ws + ((size_t)84 << 20));

  const dim3 gproj(32, 16);
  const dim3 blk(128);
  gemm_proj<<<gproj, blk, 0, stream>>>(query, Wq, bq, INV_SQRT_DK, 0, qh, nullptr);
  gemm_proj<<<gproj, blk, 0, stream>>>(key_,  Wk, bk, 1.0f,        0, kh, nullptr);
  gemm_proj<<<gproj, blk, 0, stream>>>(value, Wv, bv, 1.0f,        0, vh, nullptr);
  scores_qk<<<BB * HH * 16 * 16, blk, 0, stream>>>(qh, kh, relb, S);
  scores_bp<<<TT, blk, 0, stream>>>(qh, pos_k, S);
  softmax_rows<<<BB * HH * TT, 256, 0, stream>>>(S, mask, attn);
  attn_v_kernel<<<BB * HH * 16, blk, 0, stream>>>(attn, vh, xacc);
  attn_posv_kernel<<<TT, blk, 0, stream>>>(attn, pos_v, xacc);
  gemm_proj<<<gproj, blk, 0, stream>>>(xacc, Wo, bo, 1.0f, 1, nullptr, (float*)d_out);
}
